// ClassifierModel_33200097198456
// MI455X (gfx1250) — matrix operand order repaired and checked by reading, not compiled, not run on hardware
//
#include <hip/hip_runtime.h>
#include <hip/hip_bf16.h>

typedef __attribute__((ext_vector_type(16))) __bf16 v16bf;
typedef __attribute__((ext_vector_type(8)))  __bf16 v8bf;
typedef __attribute__((ext_vector_type(8)))  float  v8f;

static constexpr int kB   = 16;      // batch
static constexpr int kHW  = 1568;    // H*W = 224*7
static constexpr int kM1  = 25088;   // B*H*W
static constexpr int kC   = 512;
static constexpr int kD   = 256;
static constexpr int kF   = 401408;  // HW*D, flat features per batch
static constexpr int kP   = 32;
static constexpr int kKCH = 4096;    // K chunk for FC stage
static constexpr int kNKC = 98;      // kF / kKCH
static constexpr int kNT  = 12;      // 4 cls tiles (64 cols) + 8 bbox tiles (128 cols)

// ---------------------------------------------------------------------------
// Stage 1: h = relu(features @ conv_w + conv_b), stored as bf16.
// One wave per 16x16 output tile; K loop 512/32. bf16 WMMA, f32 accumulate.
// ---------------------------------------------------------------------------
__global__ __launch_bounds__(32) void conv_relu_wmma(
    const float* __restrict__ feat, const float* __restrict__ cw,
    const float* __restrict__ cb, __bf16* __restrict__ hout)
{
  const int mBase = blockIdx.x * 16;          // 0..25072 (never crosses batch: 1568%16==0)
  const int nBase = blockIdx.y * 16;          // 0..240
  const int lane  = threadIdx.x;
  const int half  = lane >> 4;
  const int i16   = lane & 15;

  const float* arow = feat + (size_t)(mBase + i16) * kC;   // A row (M = mBase+i16)
  const float* bcol = cw + nBase + i16;                    // B column (N = nBase+i16)

  v8f acc = {};
#pragma unroll 4
  for (int kb = 0; kb < kC; kb += 32) {
    // A layout: lane half selects K runs [8h..8h+7] and [16+8h..16+8h+7]
    const float4* pa0 = reinterpret_cast<const float4*>(arow + kb + 8 * half);
    const float4* pa1 = reinterpret_cast<const float4*>(arow + kb + 16 + 8 * half);
    float4 f0 = pa0[0], f1 = pa0[1], f2 = pa1[0], f3 = pa1[1];
    float af[16] = {f0.x, f0.y, f0.z, f0.w, f1.x, f1.y, f1.z, f1.w,
                    f2.x, f2.y, f2.z, f2.w, f3.x, f3.y, f3.z, f3.w};
    v16bf a;
#pragma unroll
    for (int i = 0; i < 16; ++i) a[i] = (__bf16)af[i];

    // B layout: lane half selects K runs [8h..8h+7] and [16+8h..16+8h+7]
    v16bf b;
#pragma unroll
    for (int i = 0; i < 16; ++i)
      b[i] = (__bf16)bcol[(size_t)(kb + 8 * half + i + (i >= 8 ? 8 : 0)) * kD];

    acc = __builtin_amdgcn_wmma_f32_16x16x32_bf16(false, a, false, b,
                                                  (short)0, acc, false, false);
  }

  // Epilogue: bias + relu, write bf16 h. C/D layout: VGPR r -> M = r + 8*half, N = i16.
  const float bias = cb[nBase + i16];
  const int bIdx = mBase / kHW;
#pragma unroll
  for (int r = 0; r < 8; ++r) {
    const int m = mBase + r + 8 * half;
    float v = acc[r] + bias;
    v = v > 0.f ? v : 0.f;
    const int hw = m - bIdx * kHW;
    hout[(size_t)bIdx * kF + (size_t)hw * kD + nBase + i16] = (__bf16)v;
  }
}

// ---------------------------------------------------------------------------
// Stage 2: partial [16 x 16] products of h(16 x 401408) against cls_w / bbox_w
// column tiles. Weights are streamed once (308 MB) -> non-temporal loads +
// prefetch; h (bf16, 12.8 MB) stays L2-resident across its 12 re-reads.
// Partials written to workspace (deterministic, no fp atomics).
// ---------------------------------------------------------------------------
__global__ __launch_bounds__(32) void fc_partials_wmma(
    const __bf16* __restrict__ hmat, const float* __restrict__ clsW,
    const float* __restrict__ bboxW, float* __restrict__ part)
{
  const int ntile = blockIdx.x;   // 0..11
  const int kc    = blockIdx.y;   // 0..97
  const int lane  = threadIdx.x;
  const int half  = lane >> 4;
  const int i16   = lane & 15;

  const float* wsrc;
  int ld;
  if (ntile < 4) { wsrc = clsW  + ntile * 16;        ld = 64;  }
  else           { wsrc = bboxW + (ntile - 4) * 16;  ld = 128; }
  const float* wcol = wsrc + i16;                    // B column N = i16

  const __bf16* hrow = hmat + (size_t)i16 * kF;      // A row M = batch = i16
  const int k0 = kc * kKCH;

  v8f acc = {};
#pragma unroll 2
  for (int kb = k0; kb < k0 + kKCH; kb += 32) {
    // A: two 16B bf16 runs per lane
    v8bf r0 = *reinterpret_cast<const v8bf*>(hrow + kb + 8 * half);
    v8bf r1 = *reinterpret_cast<const v8bf*>(hrow + kb + 16 + 8 * half);
    v16bf a = __builtin_shufflevector(r0, r1, 0, 1, 2, 3, 4, 5, 6, 7,
                                      8, 9, 10, 11, 12, 13, 14, 15);
    // B: streamed weights, non-temporal (do not pollute L2 where h lives)
    v16bf b;
#pragma unroll
    for (int i = 0; i < 16; ++i) {
      float wv = __builtin_nontemporal_load(wcol + (size_t)(kb + 8 * half + i + (i >= 8 ? 8 : 0)) * ld);
      b[i] = (__bf16)wv;
    }
    // stream prefetch ~128KB-256KB ahead (speculative; dropped if unmapped)
    __builtin_prefetch(wcol + (size_t)(kb + 512) * ld, 0, 1);

    acc = __builtin_amdgcn_wmma_f32_16x16x32_bf16(false, a, false, b,
                                                  (short)0, acc, false, false);
  }

  float* pdst = part + (size_t)(ntile * kNKC + kc) * 256;
#pragma unroll
  for (int r = 0; r < 8; ++r)
    pdst[(r + 8 * half) * 16 + i16] = acc[r];
}

// ---------------------------------------------------------------------------
// Stage 3: reduce 98 K-chunk partials, add biases, softmax/bbox decode.
// One block per batch image.
// ---------------------------------------------------------------------------
__global__ __launch_bounds__(192) void head_kernel(
    const float* __restrict__ part, const float* __restrict__ roi,
    const float* __restrict__ clsB, const float* __restrict__ bboxB,
    float* __restrict__ out)
{
  const int b = blockIdx.x;
  const int c = threadIdx.x;                // 0..191: [0,64)=cls col, [64,192)=bbox col
  __shared__ float sc[192];

  const int ntile = c >> 4, col = c & 15;
  float sum = (c < 64) ? clsB[c] : bboxB[c - 64];
  for (int kc = 0; kc < kNKC; ++kc)
    sum += part[(size_t)(ntile * kNKC + kc) * 256 + b * 16 + col];
  sc[c] = sum;
  __syncthreads();

  if (c < kP) {
    const int p = c;
    const float obj0 = sc[p];
    const float obj1 = sc[kP + p];
    const float objness = 1.f / (1.f + expf(obj0 - obj1));   // softmax([o0,o1])[1]

    const float bb1 = sc[64 + 32 + p];   // ref overwrites x: only bb[:,:,1] used
    float bb2 = sc[64 + 64 + p];
    float bb3 = sc[64 + 96 + p];
    bb2 = fminf(fmaxf(bb2, -10.f), 10.f);
    bb3 = fminf(fmaxf(bb3, -10.f), 10.f);

    const float* r4 = roi + ((size_t)b * kP + p) * 4;
    const float r0 = r4[0] * 16.f, r1 = r4[1] * 16.f;
    const float r2 = r4[2] * 16.f, r3 = r4[3] * 16.f;

    float* o = out + ((size_t)b * kP + p) * 5;
    o[0] = r0 - bb1 * r3;
    o[1] = r1;
    o[2] = r2 * expf(bb2);
    o[3] = r3 * expf(bb3);
    o[4] = objness;
  }
}

// ---------------------------------------------------------------------------
extern "C" void kernel_launch(void* const* d_in, const int* in_sizes, int n_in,
                              void* d_out, int out_size, void* d_ws, size_t ws_size,
                              hipStream_t stream) {
  const float* feat  = (const float*)d_in[0];
  const float* roi   = (const float*)d_in[1];
  const float* convW = (const float*)d_in[2];
  const float* convB = (const float*)d_in[3];
  const float* clsW  = (const float*)d_in[4];
  const float* clsB  = (const float*)d_in[5];
  const float* bboxW = (const float*)d_in[6];
  const float* bboxB = (const float*)d_in[7];
  float* out = (float*)d_out;

  // Workspace layout: [ h: 16*401408 bf16 = 12.85 MB ][ partials: 12*98*256 f32 = 1.2 MB ]
  __bf16* hbuf = (__bf16*)d_ws;
  float*  part = (float*)((char*)d_ws + (size_t)kB * kF * sizeof(__bf16));

  conv_relu_wmma<<<dim3(kM1 / 16, kD / 16), 32, 0, stream>>>(feat, convW, convB, hbuf);
  fc_partials_wmma<<<dim3(kNT, kNKC), 32, 0, stream>>>(hbuf, clsW, bboxW, part);
  head_kernel<<<kB, 192, 0, stream>>>(part, roi, clsB, bboxB, out);
}
